// VectorQuantizer_62904091017602
// MI455X (gfx1250) — matrix operand order repaired and checked by reading, not compiled, not run on hardware
//
#include <hip/hip_runtime.h>
#include <hip/hip_bf16.h>

#define N_E   8192
#define E_DIM 256
#define N_TOK 65536

typedef __attribute__((ext_vector_type(16))) __bf16 v16bf;
typedef __attribute__((ext_vector_type(8)))  float  v8f;

union BFrag { uint4 q[2]; v16bf v; };

// ---------------------------------------------------------------------------
// Kernel 1: codebook prep.  One wave per code row (8 rows per 256-thr block).
// Produces bf16 hi/lo split of codebook + 0.5*||c||^2 per code.
// ---------------------------------------------------------------------------
__global__ void vq_prep_codebook(const float* __restrict__ cb,
                                 __bf16* __restrict__ chi,
                                 __bf16* __restrict__ clo,
                                 float* __restrict__ cnh) {
    const int wave = threadIdx.x >> 5;
    const int lane = threadIdx.x & 31;
    const int row  = blockIdx.x * 8 + wave;

    const float* src = cb + (size_t)row * E_DIM + lane * 8;
    float4 a = ((const float4*)src)[0];
    float4 b = ((const float4*)src)[1];
    float v[8] = {a.x, a.y, a.z, a.w, b.x, b.y, b.z, b.w};

    __bf16 h[8], l[8];
    float ss = 0.0f;
#pragma unroll
    for (int i = 0; i < 8; ++i) {
        __bf16 hh = (__bf16)v[i];
        h[i] = hh;
        l[i] = (__bf16)(v[i] - (float)hh);
        ss += v[i] * v[i];
    }
#pragma unroll
    for (int off = 16; off >= 1; off >>= 1)
        ss += __shfl_xor(ss, off, 32);
    if (lane == 0) cnh[row] = 0.5f * ss;

    *(uint4*)(chi + (size_t)row * E_DIM + lane * 8) = *(uint4*)h;
    *(uint4*)(clo + (size_t)row * E_DIM + lane * 8) = *(uint4*)l;
}

// ---------------------------------------------------------------------------
// Kernel 2: WMMA GEMM + running argmax.  Codebook tiles stream to LDS via
// GLOBAL_LOAD_ASYNC_TO_LDS_B128 (ASYNCcnt), double-buffered: zero VGPR
// staging cost, one full tile of prefetch distance.
// ---------------------------------------------------------------------------
#define NT        32                    // codes per LDS tile
#define NTILES    (N_E / NT)
#define ROWPAD    264                   // bf16 elems: 256 + 8 pad
#define ROWBYTES  (ROWPAD * 2)          // 528
#define HALF_B    (NT * ROWBYTES)       // 16896: one hi (or lo) array
#define BUF_B     (2 * HALF_B)          // 33792: hi+lo for one tile
#define SMEM_B    (2 * BUF_B)           // 67584: double buffer

__device__ __forceinline__ void vq_async_ld16(unsigned lds_off, const void* g) {
    asm volatile("global_load_async_to_lds_b128 %0, %1, off"
                 :: "v"(lds_off), "v"(g) : "memory");
}

// Issue one tile's staging: 8 threads/row, 64B chunk each, hi + lo = 8 loads.
__device__ __forceinline__ void vq_stage_tile(const __bf16* __restrict__ chi,
                                              const __bf16* __restrict__ clo,
                                              unsigned lds_base, int cbase,
                                              int srow, int schnk) {
    const char* gh = (const char*)(chi + (size_t)(cbase + srow) * E_DIM) + schnk * 64;
    const char* gl = (const char*)(clo + (size_t)(cbase + srow) * E_DIM) + schnk * 64;
    const unsigned dh = lds_base + srow * ROWBYTES + schnk * 64;
    const unsigned dl = dh + HALF_B;
#pragma unroll
    for (int j = 0; j < 4; ++j) {
        vq_async_ld16(dh + j * 16, gh + j * 16);
        vq_async_ld16(dl + j * 16, gl + j * 16);
    }
}

__global__ __launch_bounds__(256) void vq_argmin(
        const float*  __restrict__ z,
        const __bf16* __restrict__ chi,
        const __bf16* __restrict__ clo,
        const float*  __restrict__ cnh,
        int*          __restrict__ best) {
    extern __shared__ char smem[];
    // Low 32 bits of a flat shared pointer == LDS byte offset (ISA 10.2).
    const unsigned smem_lds = (unsigned)(uintptr_t)&smem[0];

    const int tid  = threadIdx.x;
    const int wave = tid >> 5;
    const int lane = tid & 31;
    const int half = lane >> 4;          // 0: lanes 0-15, 1: lanes 16-31
    const int lc   = lane & 15;

    const int rowbase = blockIdx.x * 128 + wave * 16;

    // Staging role: 8 threads per code row, 64B chunk per thread.
    const int srow  = tid >> 3;
    const int schnk = tid & 7;

    // Kick off tiles 0 and 1 before the (long) A-fragment build.
    vq_stage_tile(chi, clo, smem_lds + 0 * BUF_B, 0 * NT, srow, schnk);
    vq_stage_tile(chi, clo, smem_lds + 1 * BUF_B, 1 * NT, srow, schnk);

    // ---- Load this wave's 16x256 z tile into WMMA A fragments (hi/lo) ----
    // A k order used here (same as B): lanes0-15 K=0..15, lanes16-31 K=16..31
    v16bf ahi[8], alo[8];
    {
        const int arow  = rowbase + lc;
        const int koff0 = half ? 16 : 0;
#pragma unroll
        for (int b = 0; b < 8; ++b) {
            const float* zr = z + (size_t)arow * E_DIM + b * 32 + koff0;
            float4 f0 = ((const float4*)zr)[0];
            float4 f1 = ((const float4*)(zr + 4))[0];
            float4 f2 = ((const float4*)(zr + 8))[0];
            float4 f3 = ((const float4*)(zr + 12))[0];
            float vals[16] = {f0.x, f0.y, f0.z, f0.w, f1.x, f1.y, f1.z, f1.w,
                              f2.x, f2.y, f2.z, f2.w, f3.x, f3.y, f3.z, f3.w};
#pragma unroll
            for (int e = 0; e < 16; ++e) {
                __bf16 hh = (__bf16)vals[e];
                ahi[b][e] = hh;
                alo[b][e] = (__bf16)(vals[e] - (float)hh);
            }
        }
    }

    float bv[8];
    int   bi[8];
#pragma unroll
    for (int i = 0; i < 8; ++i) { bv[i] = -__builtin_inff(); bi[i] = 0; }

    for (int ct = 0; ct < NTILES; ++ct) {
        const int cbase = ct * NT;
        // Tile ct's 8 loads done when <=8 remain (those are tile ct+1's).
        if (ct + 1 < NTILES) {
            asm volatile("s_wait_asynccnt 0x8" ::: "memory");
        } else {
            asm volatile("s_wait_asynccnt 0x0" ::: "memory");
        }
        __syncthreads();                 // all waves' tile-ct data in LDS

        const __bf16* Bh = (const __bf16*)(smem + (ct & 1) * BUF_B);
        const __bf16* Bl = (const __bf16*)(smem + (ct & 1) * BUF_B + HALF_B);
        const float cn01[2] = { cnh[cbase + lc], cnh[cbase + 16 + lc] };

#pragma unroll
        for (int s = 0; s < 2; ++s) {    // two 16-code subtiles
            v8f acc = {0.f, 0.f, 0.f, 0.f, 0.f, 0.f, 0.f, 0.f};
            const int crow = s * 16 + lc;        // B col = lane&15
#pragma unroll
            for (int b = 0; b < 8; ++b) {
                // B k order used here (same as A): lanes0-15 K=kb+0..15, lanes16-31 K=kb+16..31
                const int ko = b * 32 + (half ? 16 : 0);
                BFrag bh, bl;
                const uint4* ph = (const uint4*)(Bh + crow * ROWPAD + ko);
                const uint4* pl = (const uint4*)(Bl + crow * ROWPAD + ko);
                bh.q[0] = ph[0]; bh.q[1] = ph[1];
                bl.q[0] = pl[0]; bl.q[1] = pl[1];
                acc = __builtin_amdgcn_wmma_f32_16x16x32_bf16(
                          false, ahi[b], false, bh.v, (short)0, acc, false, false);
                acc = __builtin_amdgcn_wmma_f32_16x16x32_bf16(
                          false, alo[b], false, bh.v, (short)0, acc, false, false);
                acc = __builtin_amdgcn_wmma_f32_16x16x32_bf16(
                          false, ahi[b], false, bl.v, (short)0, acc, false, false);
            }
            const int code = cbase + s * 16 + lc;
            const float cn = cn01[s];
            // C layout: acc[i] = (M = i + 8*half, N = lc)
#pragma unroll
            for (int i = 0; i < 8; ++i) {
                float sc = acc[i] - cn;
                if (sc > bv[i]) { bv[i] = sc; bi[i] = code; }
            }
        }

        __syncthreads();                 // readers of buf[ct&1] done
        if (ct + 2 < NTILES) {           // stage tile ct+2 into buf[ct&1]
            vq_stage_tile(chi, clo, smem_lds + (ct & 1) * BUF_B,
                          cbase + 2 * NT, srow, schnk);
        }
    }

    // ---- Cross-lane argmax within each 16-lane half (N dimension) ----
#pragma unroll
    for (int i = 0; i < 8; ++i) {
        float v = bv[i];
        int   x = bi[i];
#pragma unroll
        for (int m = 1; m <= 8; m <<= 1) {
            float ov = __shfl_xor(v, m, 32);
            int   ox = __shfl_xor(x, m, 32);
            if (ov > v || (ov == v && ox < x)) { v = ov; x = ox; }
        }
        if (lc == 0) best[rowbase + i + 8 * half] = x;
    }
}

// ---------------------------------------------------------------------------
// Kernel 3: gather z_q = codebook[best[n]].  One wave per token row (1KB).
// ---------------------------------------------------------------------------
__global__ void vq_gather(const float* __restrict__ cb,
                          const int* __restrict__ best,
                          float* __restrict__ out) {
    const int wave = threadIdx.x >> 5;
    const int lane = threadIdx.x & 31;
    const int n    = blockIdx.x * 8 + wave;
    const int idx  = best[n];
    const float4* s = (const float4*)(cb + (size_t)idx * E_DIM);
    float4*       d = (float4*)(out + (size_t)n * E_DIM);
    d[lane]      = s[lane];
    d[lane + 32] = s[lane + 32];
}

// ---------------------------------------------------------------------------
extern "C" void kernel_launch(void* const* d_in, const int* in_sizes, int n_in,
                              void* d_out, int out_size, void* d_ws, size_t ws_size,
                              hipStream_t stream) {
    const float* z  = (const float*)d_in[0];   // [N_TOK, E_DIM] f32
    const float* cb = (const float*)d_in[1];   // [N_E,  E_DIM] f32
    float* out = (float*)d_out;                // [N_TOK, E_DIM] f32

    char* w = (char*)d_ws;
    __bf16* chi = (__bf16*)w;                                     // 4 MB
    __bf16* clo = (__bf16*)(w + (size_t)N_E * E_DIM * 2);         // 4 MB
    float*  cnh = (float*)(w + (size_t)N_E * E_DIM * 4);          // 32 KB
    int*    best = (int*)(w + (size_t)N_E * E_DIM * 4 + N_E * 4); // 256 KB

    vq_prep_codebook<<<N_E / 8, 256, 0, stream>>>(cb, chi, clo, cnh);
    vq_argmin<<<N_TOK / 128, 256, SMEM_B, stream>>>(z, chi, clo, cnh, best);
    vq_gather<<<N_TOK / 8, 256, 0, stream>>>(cb, best, out);
}
